// VoxelBackBone8xBIL_18459769438405
// MI455X (gfx1250) — compile-verified, matrix operand order checked by reading, not run on hardware
//
#include <hip/hip_runtime.h>
#include <hip/hip_bf16.h>

// ---------------------------------------------------------------------------
// VoxelBackBone8x on gfx1250: dense implicit-GEMM sparse-conv backbone.
// Activations stored bf16 (halve HBM traffic @ 23.3 TB/s), f32 accumulate via
// v_wmma_f32_16x16x32_bf16. Weight panels are DMA'd into LDS by the Tensor
// Data Mover (tensor_load_to_lds + s_wait_tensorcnt), overlapping the A-tile
// gather. Per layer: pack weights -> (mask dilate) -> implicit-GEMM conv ->
// masked BN stats -> normalize+ReLU.
// ---------------------------------------------------------------------------

typedef __attribute__((ext_vector_type(16))) __bf16 v16bf;
typedef __attribute__((ext_vector_type(8)))  float  v8f;
typedef unsigned int u32x4 __attribute__((ext_vector_type(4)));
typedef unsigned int u32x8 __attribute__((ext_vector_type(8)));

union FragBF { v16bf v; uint4 q[2]; };

__device__ __forceinline__ unsigned short f2bf(float f) {
    unsigned int u = __float_as_uint(f);
    unsigned int r = u + 0x7FFFu + ((u >> 16) & 1u);   // round-to-nearest-even
    return (unsigned short)(r >> 16);
}

// ---------------------------------------------------------------------------
// TDM: issue a 2D tile load (nRows x tileW bf16 elems, row stride strideElems)
// from global into LDS at lds_off. D# groups 0/1 built per CDNA5 ISA:
//   g0 = {count=1, lds_addr, global_addr[56:0], type=2}
//   g1 = {data_size=2B, tensor_dim0/1, tile_dim0/1, tensor_dim0_stride}
// OOB reads (tensorDim0 < tileW on the remainder step) zero-fill LDS, which
// keeps the LDS pitch constant at tileW.
// ---------------------------------------------------------------------------
__device__ __forceinline__ void tdm_load_panel(unsigned lds_off,
                                               const unsigned short* gptr,
                                               unsigned tensorDim0,
                                               unsigned nRows,
                                               unsigned tileW,
                                               unsigned long long strideElems) {
    unsigned long long ga = (unsigned long long)(uintptr_t)gptr;
    u32x4 g0;
    g0[0] = 1u;                                            // count=1 (valid user D#)
    g0[1] = lds_off;                                       // lds_addr
    g0[2] = (unsigned)ga;                                  // global_addr[31:0]
    g0[3] = (unsigned)((ga >> 32) & 0x01FFFFFFu) | (2u << 30);  // addr[56:32], type=2
    u32x8 g1;
    g1[0] = 0x00010000u;                                   // data_size=1 (2 bytes)
    g1[1] = (tensorDim0 & 0xFFFFu) << 16;                  // tensor_dim0[15:0]
    g1[2] = (tensorDim0 >> 16) | ((nRows & 0xFFFFu) << 16);// dim0[31:16] | tensor_dim1[15:0]
    g1[3] = (nRows >> 16) | ((tileW & 0xFFFFu) << 16);     // dim1[31:16] | tile_dim0
    g1[4] = nRows & 0xFFFFu;                               // tile_dim1 (tile_dim2=0)
    g1[5] = (unsigned)strideElems;                         // tensor_dim0_stride[31:0]
    g1[6] = (unsigned)((strideElems >> 32) & 0xFFFFu);     // stride[47:32]
    g1[7] = 0u;
    asm volatile("tensor_load_to_lds %0, %1" :: "s"(g0), "s"(g1) : "memory");
}

// ---------------------------------------------------------------------------
// zero: grid-stride word fill
// ---------------------------------------------------------------------------
__global__ void zero_kernel(unsigned int* __restrict__ p, long n) {
    long i  = (long)blockIdx.x * blockDim.x + threadIdx.x;
    long st = (long)gridDim.x * blockDim.x;
    for (; i < n; i += st) p[i] = 0u;
}

// ---------------------------------------------------------------------------
// scatter voxel features into dense bf16 grid [D,H,W,Cin] + mask
// ---------------------------------------------------------------------------
__global__ void scatter_kernel(const float* __restrict__ feats,
                               const int* __restrict__ coords,
                               unsigned short* __restrict__ x0,
                               float* __restrict__ mask0,
                               int nvox, int H, int W, int Cin) {
    int i = blockIdx.x * blockDim.x + threadIdx.x;
    if (i >= nvox) return;
    int z = coords[i * 4 + 1];
    int y = coords[i * 4 + 2];
    int x = coords[i * 4 + 3];
    long pos = ((long)z * H + y) * W + x;
    mask0[pos] = 1.0f;
    for (int c = 0; c < Cin; ++c)
        x0[pos * Cin + c] = f2bf(feats[i * Cin + c]);
}

// ---------------------------------------------------------------------------
// weight pack: OIDHW f32 -> oc-major bf16 panel wT[oc][K = off*IC+ic]
// (oc-major so the TDM 2D tile lands in LDS as [oc][32] -- already transposed)
// ---------------------------------------------------------------------------
__global__ void pack_w_kernel(const float* __restrict__ w,
                              unsigned short* __restrict__ wT,
                              int OC, int IC, int KD, int KH, int KW, int Ktot) {
    int idx = blockIdx.x * blockDim.x + threadIdx.x;
    int total = Ktot * OC;
    if (idx >= total) return;
    int k   = idx % Ktot;
    int oc  = idx / Ktot;
    int ic  = k % IC;
    int off = k / IC;
    int kw  = off % KW;
    int r   = off / KW;
    int kh  = r % KH;
    int kd  = r / KH;
    float f = w[((((long)oc * IC + ic) * KD + kd) * KH + kh) * KW + kw];
    wT[idx] = f2bf(f);
}

// ---------------------------------------------------------------------------
// mask dilation for strided ('sp') layers: mask_out = conv(mask, ones) > 0
// ---------------------------------------------------------------------------
__global__ void mask_conv_kernel(const float* __restrict__ mIn,
                                 float* __restrict__ mOut, long NposO,
                                 int Di, int Hi, int Wi, int Ho, int Wo,
                                 int KD, int KH, int KW,
                                 int sz, int sy, int sx,
                                 int pz, int py, int px) {
    long m = (long)blockIdx.x * blockDim.x + threadIdx.x;
    if (m >= NposO) return;
    int xo = (int)(m % Wo);
    long t = m / Wo;
    int yo = (int)(t % Ho);
    int zo = (int)(t / Ho);
    float v = 0.0f;
    for (int kd = 0; kd < KD; ++kd) {
        int zi = zo * sz - pz + kd;
        if ((unsigned)zi >= (unsigned)Di) continue;
        for (int kh = 0; kh < KH; ++kh) {
            int yi = yo * sy - py + kh;
            if ((unsigned)yi >= (unsigned)Hi) continue;
            for (int kw = 0; kw < KW; ++kw) {
                int xi = xo * sx - px + kw;
                if ((unsigned)xi >= (unsigned)Wi) continue;
                if (mIn[((long)zi * Hi + yi) * Wi + xi] > 0.0f) v = 1.0f;
            }
        }
    }
    mOut[m] = v;
}

// ---------------------------------------------------------------------------
// implicit-GEMM conv, bf16 WMMA, f32 accumulate.
//   M tile = 128 output voxels / block (8 waves, 16 rows/wave)
//   N      = OC = NCH*16 column chunks, accumulators resident in VGPRs
//   K      = IC*kvol, stepped by 32; A gather staged in LDS, B panel DMA'd
//            into LDS by the TDM (issued by wave 0, overlapped with A stage)
// ---------------------------------------------------------------------------
template <int NCH>
__global__ __launch_bounds__(256, 2) void conv_wmma_kernel(
    const unsigned short* __restrict__ xin,   // bf16 dense [Di,Hi,Wi,IC]
    const unsigned short* __restrict__ wT,    // bf16 [OC][Ktot] oc-major
    float* __restrict__ y,                    // f32 dense [NposO, OC]
    long NposO, int Di, int Hi, int Wi, int Ho, int Wo,
    int IC, int icShift, int KD, int KH, int KW,
    int sz, int sy, int sx, int pz, int py, int px, int Ktot) {
    constexpr int OC = NCH * 16;
    __shared__ unsigned short sA[128 * 32];   // 8 KB: 128 rows x 32 K
    __shared__ unsigned short sB[128 * 32];   // 8 KB: [oc][32 K] (TDM target)

    const int tid  = threadIdx.x;
    const int wave = tid >> 5;
    const int lane = tid & 31;
    const long mbase = (long)blockIdx.x * 128;
    const unsigned sb_base = (unsigned)(uintptr_t)(&sB[0]);

    // staging assignment: thread -> (row = tid/2, k-half = 16*(tid&1))
    const int arow = tid >> 1;
    const int kh0  = (tid & 1) << 4;
    const long m   = mbase + arow;
    const bool mval = (m < NposO);
    const long mm  = mval ? m : 0;
    const int xo = (int)(mm % Wo);
    const long t1 = mm / Wo;
    const int yo = (int)(t1 % Ho);
    const int zo = (int)(t1 / Ho);
    const int icMask = IC - 1;
    const int KHW = KH * KW;

    v8f acc[NCH] = {};

    for (int k0 = 0; k0 < Ktot; k0 += 32) {
        // ---- issue B-panel DMA (TDM): OC rows x 32 elems, row stride Ktot ----
        if (wave == 0) {
            tdm_load_panel(sb_base, wT + (long)k0,
                           (unsigned)(Ktot - k0),        // tensor_dim0: OOB -> zero fill
                           (unsigned)OC, 32u,
                           (unsigned long long)Ktot);
        }
        if (k0 + 32 < Ktot)
            __builtin_prefetch(&wT[k0 + 32], 0, 1);       // global_prefetch_b8 (L2 warm)

        // ---- stage A (gather + bounds/zero fill), overlapped with the DMA ----
        if (icShift >= 4) {
            // IC in {16,32,64}: this thread's 16 K-elems lie in ONE kernel
            // offset and are contiguous channels -> two aligned b128 loads.
            int k = k0 + kh0;
            uint4 lo = make_uint4(0, 0, 0, 0), hi = make_uint4(0, 0, 0, 0);
            if (mval && k < Ktot) {
                int off = k >> icShift;
                int ic0 = k & icMask;
                int kd  = off / KHW;
                int rr  = off - kd * KHW;
                int kh  = rr / KW;
                int kw  = rr - kh * KW;
                int zi = zo * sz - pz + kd;
                int yi = yo * sy - py + kh;
                int xi = xo * sx - px + kw;
                if ((unsigned)zi < (unsigned)Di && (unsigned)yi < (unsigned)Hi &&
                    (unsigned)xi < (unsigned)Wi) {
                    const uint4* g = reinterpret_cast<const uint4*>(
                        &xin[(((long)zi * Hi + yi) * Wi + xi) * IC + ic0]);
                    lo = g[0];
                    hi = g[1];
                }
            }
            uint4* sp = reinterpret_cast<uint4*>(&sA[arow * 32 + kh0]);
            sp[0] = lo;
            sp[1] = hi;
        } else {
            // IC = 4 (first layer): per-element gather
            #pragma unroll
            for (int j = 0; j < 16; ++j) {
                int k = k0 + kh0 + j;
                unsigned short v = 0;
                if (mval && (k < Ktot)) {
                    int off = k >> icShift;
                    int ic  = k & icMask;
                    int kd  = off / KHW;
                    int rr  = off - kd * KHW;
                    int kh  = rr / KW;
                    int kw  = rr - kh * KW;
                    int zi = zo * sz - pz + kd;
                    int yi = yo * sy - py + kh;
                    int xi = xo * sx - px + kw;
                    if ((unsigned)zi < (unsigned)Di && (unsigned)yi < (unsigned)Hi &&
                        (unsigned)xi < (unsigned)Wi) {
                        v = xin[(((long)zi * Hi + yi) * Wi + xi) * IC + ic];
                    }
                }
                sA[arow * 32 + kh0 + j] = v;
            }
        }

        // publish: wave 0 drains the DMA, then workgroup barrier
        if (wave == 0) __builtin_amdgcn_s_wait_tensorcnt(0);
        __syncthreads();

        // ---- fragments + WMMA ----
        FragBF a;
        {
            int row = (wave << 4) + (lane & 15);
            const uint4* p = reinterpret_cast<const uint4*>(
                &sA[row * 32 + ((lane >> 4) << 4)]);
            a.q[0] = p[0];
            a.q[1] = p[1];
        }
        #pragma unroll
        for (int c = 0; c < NCH; ++c) {
            FragBF b;
            int oc = (c << 4) + (lane & 15);
            const uint4* p = reinterpret_cast<const uint4*>(
                &sB[oc * 32 + ((lane >> 4) << 4)]);
            b.q[0] = p[0];
            b.q[1] = p[1];
            acc[c] = __builtin_amdgcn_wmma_f32_16x16x32_bf16(
                false, a.v, false, b.v, (short)0, acc[c], false, false);
        }
        __syncthreads();   // all readers done before next TDM overwrites sB
    }

    // ---- store C (16x16 f32 layout: lanes 0-15 rows r, lanes 16-31 rows r+8) ----
    const int rbase = (wave << 4) + ((lane >> 4) << 3);
    #pragma unroll
    for (int c = 0; c < NCH; ++c) {
        int col = (c << 4) + (lane & 15);
        #pragma unroll
        for (int r = 0; r < 8; ++r) {
            long row = mbase + rbase + r;
            if (row < NposO) y[row * OC + col] = acc[c][r];
        }
    }
}

// ---------------------------------------------------------------------------
// masked BN statistics: register partial sums per (channel, pos-group), then
// one global_atomic_add_f32 per thread. stats = [count, sum[OC], sumsq[OC]].
// ---------------------------------------------------------------------------
__global__ __launch_bounds__(256) void bn_reduce_kernel(
    const float* __restrict__ y, const float* __restrict__ mask,
    float* __restrict__ stats, long Npos, int OC, int ocShift) {
    int tid  = threadIdx.x;
    int oc   = tid & (OC - 1);
    int grp  = tid >> ocShift;
    int ngrp = 256 >> ocShift;
    long pbase = (long)blockIdx.x * 256;
    long pend  = pbase + 256;
    if (pend > Npos) pend = Npos;
    float s = 0.0f, ss = 0.0f, cnt = 0.0f;
    for (long p = pbase + grp; p < pend; p += ngrp) {
        if (mask[p] > 0.0f) {
            float v = y[p * OC + oc];
            s  += v;
            ss += v * v;
            cnt += 1.0f;
        }
    }
    atomicAdd(&stats[1 + oc], s);
    atomicAdd(&stats[1 + OC + oc], ss);
    if (oc == 0) atomicAdd(&stats[0], cnt);
}

// ---------------------------------------------------------------------------
// normalize + ReLU + mask; write bf16 next-layer input (or f32 NCDHW on final)
// ---------------------------------------------------------------------------
__global__ void bn_norm_kernel(const float* __restrict__ y,
                               const float* __restrict__ mask,
                               const float* __restrict__ stats,
                               const float* __restrict__ gamma,
                               const float* __restrict__ beta,
                               unsigned short* __restrict__ xout,
                               float* __restrict__ fout,
                               long Npos, int OC, int ocShift, int isFinal) {
    long idx = (long)blockIdx.x * blockDim.x + threadIdx.x;
    long total = Npos << ocShift;
    if (idx >= total) return;
    int oc  = (int)(idx & (OC - 1));
    long pos = idx >> ocShift;
    float n = stats[0];
    n = (n < 1.0f) ? 1.0f : n;
    float mean = stats[1 + oc] / n;
    float var  = stats[1 + OC + oc] / n - mean * mean;
    float rs   = __frsqrt_rn(var + 1e-3f);
    float v = (y[idx] - mean) * rs * gamma[oc] + beta[oc];
    v = (v > 0.0f) ? v : 0.0f;
    if (!(mask[pos] > 0.0f)) v = 0.0f;
    if (isFinal) fout[(long)oc * Npos + pos] = v;
    else         xout[idx] = f2bf(v);
}

// ---------------------------------------------------------------------------
// host side
// ---------------------------------------------------------------------------
struct LSpec { int sp, oc, ic, k[3], s[3], p[3]; };
static const LSpec LS[12] = {
    {0, 16,  4, {3,3,3}, {1,1,1}, {1,1,1}},
    {0, 16, 16, {3,3,3}, {1,1,1}, {1,1,1}},
    {1, 32, 16, {3,3,3}, {2,2,2}, {1,1,1}},
    {0, 32, 32, {3,3,3}, {1,1,1}, {1,1,1}},
    {0, 32, 32, {3,3,3}, {1,1,1}, {1,1,1}},
    {1, 64, 32, {3,3,3}, {2,2,2}, {1,1,1}},
    {0, 64, 64, {3,3,3}, {1,1,1}, {1,1,1}},
    {0, 64, 64, {3,3,3}, {1,1,1}, {1,1,1}},
    {1, 64, 64, {3,3,3}, {2,2,2}, {0,1,1}},
    {0, 64, 64, {3,3,3}, {1,1,1}, {1,1,1}},
    {0, 64, 64, {3,3,3}, {1,1,1}, {1,1,1}},
    {1,128, 64, {3,1,1}, {2,1,1}, {0,0,0}},
};

static int ilog2i(int v) { int s = 0; while ((1 << s) < v) ++s; return s; }

extern "C" void kernel_launch(void* const* d_in, const int* in_sizes, int n_in,
                              void* d_out, int out_size, void* d_ws, size_t ws_size,
                              hipStream_t stream) {
    (void)in_sizes; (void)n_in; (void)out_size; (void)ws_size;
    const float* feats  = (const float*)d_in[0];
    const int*   coords = (const int*)d_in[1];
    // weights: d_in[2..13], gammas: d_in[14..25], betas: d_in[26..37]

    // ---- workspace layout ----
    const long XCAP = 3870720L * 16;       // max activation elems (bf16)
    const long MCAP = 3870720L;            // max voxel count
    const long WCAP = 2048L * 128;         // max packed weight elems
    char* base = (char*)d_ws;
    size_t off = 0;
    auto alloc = [&](size_t bytes) -> void* {
        void* p = base + off;
        off = (off + bytes + 255) & ~(size_t)255;
        return p;
    };
    unsigned short* xb[2];
    xb[0] = (unsigned short*)alloc(XCAP * 2);
    xb[1] = (unsigned short*)alloc(XCAP * 2);
    float* yb = (float*)alloc(XCAP * 4);
    float* mb[2];
    mb[0] = (float*)alloc(MCAP * 4);
    mb[1] = (float*)alloc(MCAP * 4);
    unsigned short* wpack = (unsigned short*)alloc(WCAP * 2);
    float* stats = (float*)alloc(257 * 4);

    int D = 42, H = 320, W = 288;
    const int CIN = 4, NVOX = 40000;
    long pos0 = (long)D * H * W;

    // ---- init: zero x0 + mask0, scatter voxels ----
    zero_kernel<<<4096, 256, 0, stream>>>((unsigned int*)xb[0], (pos0 * CIN + 1) / 2);
    zero_kernel<<<4096, 256, 0, stream>>>((unsigned int*)mb[0], pos0);
    scatter_kernel<<<(NVOX + 255) / 256, 256, 0, stream>>>(
        feats, coords, xb[0], mb[0], NVOX, H, W, CIN);

    int xc = 0, mc = 0;
    for (int i = 0; i < 12; ++i) {
        const LSpec& L = LS[i];
        int Do = (D + 2 * L.p[0] - L.k[0]) / L.s[0] + 1;
        int Ho = (H + 2 * L.p[1] - L.k[1]) / L.s[1] + 1;
        int Wo = (W + 2 * L.p[2] - L.k[2]) / L.s[2] + 1;
        long NposO = (long)Do * Ho * Wo;
        int kvol = L.k[0] * L.k[1] * L.k[2];
        int Ktot = L.ic * kvol;
        int icShift = ilog2i(L.ic);
        int ocShift = ilog2i(L.oc);

        // pack weights to oc-major bf16 panel
        {
            int tot = Ktot * L.oc;
            pack_w_kernel<<<(tot + 255) / 256, 256, 0, stream>>>(
                (const float*)d_in[2 + i], wpack, L.oc, L.ic,
                L.k[0], L.k[1], L.k[2], Ktot);
        }

        // mask update for strided layers
        float* mIn = mb[mc];
        float* mOut = mIn;
        if (L.sp) {
            mOut = mb[1 - mc];
            mask_conv_kernel<<<(int)((NposO + 255) / 256), 256, 0, stream>>>(
                mIn, mOut, NposO, D, H, W, Ho, Wo,
                L.k[0], L.k[1], L.k[2], L.s[0], L.s[1], L.s[2],
                L.p[0], L.p[1], L.p[2]);
            mc ^= 1;
        }

        // implicit-GEMM conv with bf16 WMMA + TDM weight staging
        int nblk = (int)((NposO + 127) / 128);
        switch (L.oc >> 4) {
            case 1:
                conv_wmma_kernel<1><<<nblk, 256, 0, stream>>>(
                    xb[xc], wpack, yb, NposO, D, H, W, Ho, Wo, L.ic, icShift,
                    L.k[0], L.k[1], L.k[2], L.s[0], L.s[1], L.s[2],
                    L.p[0], L.p[1], L.p[2], Ktot);
                break;
            case 2:
                conv_wmma_kernel<2><<<nblk, 256, 0, stream>>>(
                    xb[xc], wpack, yb, NposO, D, H, W, Ho, Wo, L.ic, icShift,
                    L.k[0], L.k[1], L.k[2], L.s[0], L.s[1], L.s[2],
                    L.p[0], L.p[1], L.p[2], Ktot);
                break;
            case 4:
                conv_wmma_kernel<4><<<nblk, 256, 0, stream>>>(
                    xb[xc], wpack, yb, NposO, D, H, W, Ho, Wo, L.ic, icShift,
                    L.k[0], L.k[1], L.k[2], L.s[0], L.s[1], L.s[2],
                    L.p[0], L.p[1], L.p[2], Ktot);
                break;
            default:
                conv_wmma_kernel<8><<<nblk, 256, 0, stream>>>(
                    xb[xc], wpack, yb, NposO, D, H, W, Ho, Wo, L.ic, icShift,
                    L.k[0], L.k[1], L.k[2], L.s[0], L.s[1], L.s[2],
                    L.p[0], L.p[1], L.p[2], Ktot);
                break;
        }

        // masked BN stats + normalize/ReLU
        zero_kernel<<<1, 256, 0, stream>>>((unsigned int*)stats, 257);
        bn_reduce_kernel<<<(int)((NposO + 255) / 256), 256, 0, stream>>>(
            yb, mOut, stats, NposO, L.oc, ocShift);
        long tote = NposO * L.oc;
        bn_norm_kernel<<<(int)((tote + 255) / 256), 256, 0, stream>>>(
            yb, mOut, stats, (const float*)d_in[14 + i], (const float*)d_in[26 + i],
            xb[1 - xc], (float*)d_out, NposO, L.oc, ocShift, (i == 11) ? 1 : 0);

        xc ^= 1;
        D = Do; H = Ho; W = Wo;
    }
}
